// CRF_31585189495099
// MI455X (gfx1250) — compile-verified, matrix operand order checked by reading, not run on hardware
//
#include <hip/hip_runtime.h>
#include <hip/hip_bf16.h>

typedef __attribute__((ext_vector_type(16))) __bf16 v16bf;
typedef __attribute__((ext_vector_type(8)))  float  v8f;

#define CRF_B 64
#define CRF_T 2048
#define CRF_D 512
#define CRF_U 32

// ---- helpers ---------------------------------------------------------------

// One-time prep path: round-to-nearest-even f32 -> bf16.
static __device__ inline __bf16 f2bf(float f) {
  union { float f; unsigned u; } in; in.f = f;
  unsigned u = in.u;
  u += 0x7FFFu + ((u >> 16) & 1u);
  union { unsigned short s; __bf16 b; } out;
  out.s = (unsigned short)(u >> 16);
  return out.b;
}

// Hot path: pack two f32 into two bf16 (truncation) with ONE v_perm_b32.
// dst[15:0] = a[31:16], dst[31:16] = b[31:16].
static __device__ inline unsigned pack2bf(float a, float b) {
  union { float f; unsigned u; } ua, ub; ua.f = a; ub.f = b;
  return __builtin_amdgcn_perm(ub.u, ua.u, 0x07060302u);
}

// ---- kernel 1: pre-convert kernel[512,32] f32 -> bf16 WMMA B fragments -----
// Layout: frag[(kstep*2 + ntile)*32 + lane][16] __bf16, contiguous per lane.
__global__ __launch_bounds__(1024)
void crf_prep_bfrag(const float* __restrict__ kern, __bf16* __restrict__ bfrag) {
  int tid   = threadIdx.x;                 // 0..1023
  int kstep = tid >> 6;                    // 0..15 (K block of 32)
  int rem   = tid & 63;
  int nt    = rem >> 5;                    // n-tile 0/1
  int lane  = rem & 31;
  int nr    = lane & 15;
  int hi    = lane >> 4;
  __bf16* dst = bfrag + (((kstep * 2 + nt) * 32 + lane) << 4);
  int kbase = kstep * 32 + hi * 8;
#pragma unroll
  for (int e = 0; e < 8; ++e)
    dst[e] = f2bf(kern[(kbase + e) * CRF_U + nt * 16 + nr]);
#pragma unroll
  for (int e = 0; e < 8; ++e)
    dst[8 + e] = f2bf(kern[(kbase + 16 + e) * CRF_U + nt * 16 + nr]);
}

// ---- kernel 2: logits = x @ kernel + bias via bf16 WMMA --------------------
// 8 waves / block, each wave: 16 rows x 32 cols, K=512 in 16 steps of 32.
__global__ __launch_bounds__(256)
void crf_logits_wmma(const float* __restrict__ x,
                     const float* __restrict__ bias,
                     const __bf16* __restrict__ bfrag,
                     float* __restrict__ logits) {
  const int lane   = threadIdx.x & 31;
  const int wave   = threadIdx.x >> 5;
  const int m_base = (blockIdx.x * 8 + wave) * 16;   // row tile
  const int mr     = lane & 15;
  const int hi     = lane >> 4;

  const float* xr = x + (size_t)(m_base + mr) * CRF_D;
  const v16bf* bf = (const v16bf*)bfrag;

  v8f c0 = {};
  v8f c1 = {};

#pragma unroll 4
  for (int ks = 0; ks < 16; ++ks) {
    const int k0 = ks * 32;
    // A fragment: 16-bit A 16x32 layout (lane hi selects K chunks {0..7,16..23}
    // vs {8..15,24..31}); 4x float4 contiguous loads, perm-pack to bf16.
    const float* p = xr + k0 + hi * 8;
    float4 q0 = *(const float4*)(p);
    float4 q1 = *(const float4*)(p + 4);
    float4 q2 = *(const float4*)(p + 16);
    float4 q3 = *(const float4*)(p + 20);
    if (ks + 1 < 16) __builtin_prefetch(p + 32, 0, 1);

    union { v16bf v; unsigned u[8]; } af;
    af.u[0] = pack2bf(q0.x, q0.y);
    af.u[1] = pack2bf(q0.z, q0.w);
    af.u[2] = pack2bf(q1.x, q1.y);
    af.u[3] = pack2bf(q1.z, q1.w);
    af.u[4] = pack2bf(q2.x, q2.y);
    af.u[5] = pack2bf(q2.z, q2.w);
    af.u[6] = pack2bf(q3.x, q3.y);
    af.u[7] = pack2bf(q3.z, q3.w);

    v16bf b0 = bf[(ks * 2 + 0) * 32 + lane];
    v16bf b1 = bf[(ks * 2 + 1) * 32 + lane];

    c0 = __builtin_amdgcn_wmma_f32_16x16x32_bf16(false, af.v, false, b0,
                                                 (short)0, c0, false, false);
    c1 = __builtin_amdgcn_wmma_f32_16x16x32_bf16(false, af.v, false, b1,
                                                 (short)0, c1, false, false);
  }

  const float bc0 = bias[mr];
  const float bc1 = bias[16 + mr];
#pragma unroll
  for (int r = 0; r < 8; ++r) {
    const size_t row = (size_t)(m_base + r + hi * 8);
    logits[row * CRF_U + mr]      = c0[r] + bc0;
    logits[row * CRF_U + 16 + mr] = c1[r] + bc1;
  }
}

// ---- kernel 3: Viterbi decode, one wave per batch --------------------------

// Uniform broadcast of lane i's value: v_readlane_b32 (VALU/SGPR, no DS traffic).
static __device__ inline float bcast_lane(float v, int i) {
  return __int_as_float(__builtin_amdgcn_readlane(__float_as_int(v), i));
}

static __device__ inline void wave_argmax(float v, int idx, float& ov_, int& oi_) {
#pragma unroll
  for (int off = 16; off > 0; off >>= 1) {
    float ov = __shfl_xor(v, off, 32);
    int   oi = __shfl_xor(idx, off, 32);
    if (ov > v || (ov == v && oi < idx)) { v = ov; idx = oi; }  // first-max tie-break
  }
  ov_ = v; oi_ = idx;
}

__global__ __launch_bounds__(32)
void crf_viterbi(const float* __restrict__ logits,
                 const int*   __restrict__ nwords,
                 const float* __restrict__ trans,
                 float* __restrict__ alphas,
                 float* __restrict__ pred,
                 float* __restrict__ score) {
  const int b = blockIdx.x;
  const int j = threadIdx.x;          // tag index (lane)

  __shared__ float st[CRF_U * 33];    // padded stride 33: conflict-free gathers
#pragma unroll
  for (int i = 0; i < CRF_U; ++i) st[i * 33 + j] = trans[i * CRF_U + j];
  __syncthreads();

  int len = nwords[b];
  if (len < 1) len = 1;
  if (len > CRF_T) len = CRF_T;

  const float* lg = logits + (size_t)b * CRF_T * CRF_U;
  float*       al = alphas + (size_t)b * CRF_T * CRF_U;
  float*       pr = pred   + (size_t)b * CRF_T;

  float tc[CRF_U];                    // trans column j in registers
#pragma unroll
  for (int i = 0; i < CRF_U; ++i) tc[i] = st[i * 33 + j];

  float alpha = lg[j];
  al[j] = alpha;

  // Forward: alpha_t[j] = max_i(alpha_{t-1}[i] + trans[i][j]) + logit_t[j]
  for (int t = 1; t < len; ++t) {
    const float lt = lg[t * CRF_U + j];
    float s[CRF_U];
#pragma unroll
    for (int i = 0; i < CRF_U; ++i)
      s[i] = bcast_lane(alpha, i) + tc[i];   // v_readlane + v_add_f32 (s,v)
#pragma unroll
    for (int stride = 16; stride > 0; stride >>= 1)
#pragma unroll
      for (int i = 0; i < 16; ++i)
        if (i < stride) s[i] = fmaxf(s[i], s[i + stride]);
    alpha = s[0] + lt;
    al[t * CRF_U + j] = alpha;
  }

  // best score / last tag
  float best; int last;
  wave_argmax(alpha, j, best, last);
  if (j == 0) score[b] = best;

  // frozen tail: pred[t] = last for t in [len-1, T)
  for (int t = len - 1 + j; t < CRF_T; t += 32) pr[t] = (float)last;

  // Backward: recompute argmax from stored alphas (no backpointer storage)
  int tag = last;
  for (int t = len - 1; t >= 1; --t) {
    const float sv = al[(t - 1) * CRF_U + j] + st[j * 33 + tag];
    float bv; int bi;
    wave_argmax(sv, j, bv, bi);
    tag = bi;
    if (j == 0) pr[t - 1] = (float)tag;
  }
}

// ---- host launch -----------------------------------------------------------

extern "C" void kernel_launch(void* const* d_in, const int* in_sizes, int n_in,
                              void* d_out, int out_size, void* d_ws, size_t ws_size,
                              hipStream_t stream) {
  const float* x      = (const float*)d_in[0];
  const int*   nwords = (const int*)d_in[1];
  const float* kern   = (const float*)d_in[2];
  const float* trans  = (const float*)d_in[3];
  const float* bias   = (const float*)d_in[4];

  const size_t n_logits = (size_t)CRF_B * CRF_T * CRF_U;      // 4,194,304 f32
  float*  logits = (float*)d_ws;
  float*  alphas = logits + n_logits;
  __bf16* bfrag  = (__bf16*)((char*)d_ws + 2 * n_logits * sizeof(float));

  float* pred  = (float*)d_out;                // [B,T] ids stored as float
  float* score = pred + (size_t)CRF_B * CRF_T; // [B]

  crf_prep_bfrag<<<1, 1024, 0, stream>>>(kern, bfrag);

  // M = B*T = 131072 rows; 16 rows/wave, 8 waves/block -> 1024 blocks
  crf_logits_wmma<<<1024, 256, 0, stream>>>(x, bias, bfrag, logits);

  crf_viterbi<<<CRF_B, 32, 0, stream>>>(logits, nwords, trans, alphas, pred, score);
}
